// GraphSage_Simple_55989193671326
// MI455X (gfx1250) — matrix operand order repaired and checked by reading, not compiled, not run on hardware
//
#include <hip/hip_runtime.h>

#define N_NODES 50000
#define N_EDGES 800000
#define D_IN    128
#define D_HID   256
#define D_OUT   128

typedef __attribute__((ext_vector_type(16))) _Float16 v16h;
typedef __attribute__((ext_vector_type(8)))  _Float16 v8h;
typedef __attribute__((ext_vector_type(4)))  _Float16 v4h;
typedef __attribute__((ext_vector_type(8)))  float    v8f;

__global__ void zero_f32(float* __restrict__ p, int n) {
  int i = blockIdx.x * blockDim.x + threadIdx.x;
  if (i < n) p[i] = 0.0f;
}

__global__ void degree_kernel(const long long* __restrict__ dst,
                              float* __restrict__ cnt, int E) {
  int e = blockIdx.x * blockDim.x + threadIdx.x;
  if (e < E) atomicAdd(&cnt[(int)dst[e]], 1.0f);
}

// 4 elements per thread: b128 load, b64 store
__global__ void f32_to_f16_v4(const float* __restrict__ in,
                              _Float16* __restrict__ out, int n4) {
  int i = blockIdx.x * blockDim.x + threadIdx.x;
  if (i < n4) {
    float4 v = ((const float4*)in)[i];
    v4h h;
    h[0] = (_Float16)v.x; h[1] = (_Float16)v.y;
    h[2] = (_Float16)v.z; h[3] = (_Float16)v.w;
    ((v4h*)out)[i] = h;
  }
}

// W: [K x C] row-major f32  ->  Wt: [C x K] row-major f16
__global__ void transpose_to_f16(const float* __restrict__ W,
                                 _Float16* __restrict__ Wt, int K, int C) {
  int i = blockIdx.x * blockDim.x + threadIdx.x;
  if (i < K * C) {
    int k = i / C, c = i % C;
    Wt[c * K + k] = (_Float16)W[i];
  }
}

// agg[dst[e]*D + d] += x[src[e]*D + d]; D = 1<<shift; one thread per (edge,dim)
__global__ void scatter_add(const float* __restrict__ x,
                            const long long* __restrict__ src,
                            const long long* __restrict__ dst,
                            float* __restrict__ agg, int E, int shift) {
  int tid = blockIdx.x * blockDim.x + threadIdx.x;
  int D = 1 << shift;
  int e = tid >> shift;
  int d = tid & (D - 1);
  if (e < E) {
    int s = (int)src[e], t = (int)dst[e];
    atomicAdd(&agg[t * D + d], x[s * D + d]);
  }
}

// 4 elements per thread; D multiple of 4 so all 4 share one row
__global__ void mean_to_f16_v4(const float* __restrict__ agg,
                               const float* __restrict__ cnt,
                               _Float16* __restrict__ mh, int N, int shift) {
  int i = blockIdx.x * blockDim.x + threadIdx.x;
  if (i < (N << shift) / 4) {
    int n = (i * 4) >> shift;
    float c = cnt[n];
    c = (c < 1.0f) ? 1.0f : c;
    float r = 1.0f / c;
    float4 v = ((const float4*)agg)[i];
    v4h h;
    h[0] = (_Float16)(v.x * r); h[1] = (_Float16)(v.y * r);
    h[2] = (_Float16)(v.z * r); h[3] = (_Float16)(v.w * r);
    ((v4h*)mh)[i] = h;
  }
}

// C = sigmoid( A1@B1t^T + A2@B2t^T + bias ), EPI=1: L2-normalize rows first.
// A*: [Nrows x K] f16 row-major. B*t: [Ncols x K] f16 row-major (transposed W).
// Block: 256 threads = 8 waves; block tile = 16 rows x 128 cols (wave = 16x16).
// K is compile-time: the K-loop fully unrolls into load-clauses + WMMAs with
// immediate offsets. A tiles are shared by all 8 waves via WGP cache (no LDS
// staging, no barriers in the main loop).
template <int K, int EPI>
__global__ __launch_bounds__(256) void sage_gemm(
    const _Float16* __restrict__ A1, const _Float16* __restrict__ B1t,
    const _Float16* __restrict__ A2, const _Float16* __restrict__ B2t,
    const float* __restrict__ bias,
    float* __restrict__ outF, _Float16* __restrict__ outH, int Ncols) {
  __shared__ float Cs[16 * 128];   // epilogue-1 staging only
  __shared__ float rnorm[16];

  const int tid   = threadIdx.x;
  const int lane  = tid & 31;
  const int wave  = tid >> 5;
  const int khalf = lane >> 4;   // 0: lanes 0-15, 1: lanes 16-31
  const int l16   = lane & 15;
  const int r0    = blockIdx.y * 16;
  const int n     = blockIdx.x * 128 + wave * 16 + l16;  // global out column

  // Per-lane base pointers; all loop accesses are base + immediate offset.
  const _Float16* pa1 = A1 + (size_t)(r0 + l16) * K + 16 * khalf;
  const _Float16* pa2 = A2 + (size_t)(r0 + l16) * K + 16 * khalf;
  const _Float16* pb1 = B1t + (size_t)n * K + 16 * khalf;
  const _Float16* pb2 = B2t + (size_t)n * K + 16 * khalf;

  float bn = bias[n];
  v8f acc;
#pragma unroll
  for (int i = 0; i < 8; ++i) acc[i] = bn;

#pragma unroll
  for (int kk = 0; kk < K; kk += 32) {
    // A 16x32 f16 fragment: lane = row l16; K chunks
    // [kk+16*khalf,+8) -> VGPRs 0-3, [kk+16*khalf+8,+8) -> VGPRs 4-7.
    // Each chunk contiguous & 16B-aligned -> global_load_b128.
    v8h a1lo = *(const v8h*)(pa1 + kk);
    v8h a1hi = *(const v8h*)(pa1 + kk + 8);
    v8h a2lo = *(const v8h*)(pa2 + kk);
    v8h a2hi = *(const v8h*)(pa2 + kk + 8);
    v16h a1 = __builtin_shufflevector(a1lo, a1hi, 0, 1, 2, 3, 4, 5, 6, 7, 8,
                                      9, 10, 11, 12, 13, 14, 15);
    v16h a2 = __builtin_shufflevector(a2lo, a2hi, 0, 1, 2, 3, 4, 5, 6, 7, 8,
                                      9, 10, 11, 12, 13, 14, 15);
    // B 32x16 f16 fragment: lane = column n, K = [kk+16*khalf, +16):
    // 16 contiguous halfs, 32B-aligned -> 2x global_load_b128.
    v16h b1 = *(const v16h*)(pb1 + kk);
    v16h b2 = *(const v16h*)(pb2 + kk);

    acc = __builtin_amdgcn_wmma_f32_16x16x32_f16(false, a1, false, b1,
                                                 (short)0, acc, false, false);
    acc = __builtin_amdgcn_wmma_f32_16x16x32_f16(false, a2, false, b2,
                                                 (short)0, acc, false, false);
  }

  if (EPI == 0) {
#pragma unroll
    for (int i = 0; i < 8; ++i) {
      int row = r0 + i + 8 * khalf;   // C layout: VGPR i -> M = i + 8*khalf
      float v = 1.0f / (1.0f + __expf(-acc[i]));
      outF[row * Ncols + n] = v;
      outH[row * Ncols + n] = (_Float16)v;
    }
  } else {
    int cl = wave * 16 + l16;  // column within 128-wide block tile
#pragma unroll
    for (int i = 0; i < 8; ++i) Cs[(i + 8 * khalf) * 128 + cl] = acc[i];
    __syncthreads();
    if (tid < 16) {
      float ss = 0.0f;
      for (int c = 0; c < 128; ++c) {
        float v = Cs[tid * 128 + c];
        ss += v * v;
      }
      rnorm[tid] = 1.0f / fmaxf(sqrtf(ss), 1e-12f);
    }
    __syncthreads();
#pragma unroll
    for (int j = 0; j < 8; ++j) {
      int idx = tid * 8 + j;
      int r = idx >> 7, c = idx & 127;
      float v = Cs[idx] * rnorm[r];
      outF[(r0 + r) * Ncols + blockIdx.x * 128 + c] =
          1.0f / (1.0f + __expf(-v));
    }
  }
}

extern "C" void kernel_launch(void* const* d_in, const int* in_sizes, int n_in,
                              void* d_out, int out_size, void* d_ws,
                              size_t ws_size, hipStream_t stream) {
  const float*     x   = (const float*)d_in[0];
  const long long* ei  = (const long long*)d_in[1];  // int64 (2, E)
  const float*     W1l = (const float*)d_in[2];
  const float*     W1r = (const float*)d_in[3];
  const float*     b1  = (const float*)d_in[4];
  const float*     W2l = (const float*)d_in[5];
  const float*     W2r = (const float*)d_in[6];
  const float*     b2  = (const float*)d_in[7];
  const long long* src = ei;
  const long long* dst = ei + N_EDGES;

  char* p = (char*)d_ws;
  auto alloc = [&](size_t bytes) -> char* {
    char* r = p;
    p += (bytes + 255) & ~(size_t)255;
    return r;
  };
  float*    cnt    = (float*)alloc((size_t)N_NODES * 4);
  float*    agg1   = (float*)alloc((size_t)N_NODES * D_IN * 4);
  float*    agg2   = (float*)alloc((size_t)N_NODES * D_HID * 4);
  float*    x1     = (float*)alloc((size_t)N_NODES * D_HID * 4);
  _Float16* xh     = (_Float16*)alloc((size_t)N_NODES * D_IN * 2);
  _Float16* mean1h = (_Float16*)alloc((size_t)N_NODES * D_IN * 2);
  _Float16* x1h    = (_Float16*)alloc((size_t)N_NODES * D_HID * 2);
  _Float16* mean2h = (_Float16*)alloc((size_t)N_NODES * D_HID * 2);
  _Float16* w1lt   = (_Float16*)alloc((size_t)D_IN * D_HID * 2);
  _Float16* w1rt   = (_Float16*)alloc((size_t)D_IN * D_HID * 2);
  _Float16* w2lt   = (_Float16*)alloc((size_t)D_HID * D_OUT * 2);
  _Float16* w2rt   = (_Float16*)alloc((size_t)D_HID * D_OUT * 2);

  const int T = 256;
  // zero accumulators (must be fresh on every graph replay)
  zero_f32<<<(N_NODES + T - 1) / T, T, 0, stream>>>(cnt, N_NODES);
  zero_f32<<<(N_NODES * D_IN + T - 1) / T, T, 0, stream>>>(agg1, N_NODES * D_IN);
  zero_f32<<<(N_NODES * D_HID + T - 1) / T, T, 0, stream>>>(agg2, N_NODES * D_HID);

  degree_kernel<<<(N_EDGES + T - 1) / T, T, 0, stream>>>(dst, cnt, N_EDGES);

  f32_to_f16_v4<<<(N_NODES * D_IN / 4 + T - 1) / T, T, 0, stream>>>(
      x, xh, N_NODES * D_IN / 4);
  transpose_to_f16<<<(D_IN * D_HID + T - 1) / T, T, 0, stream>>>(W1l, w1lt, D_IN, D_HID);
  transpose_to_f16<<<(D_IN * D_HID + T - 1) / T, T, 0, stream>>>(W1r, w1rt, D_IN, D_HID);
  transpose_to_f16<<<(D_HID * D_OUT + T - 1) / T, T, 0, stream>>>(W2l, w2lt, D_HID, D_OUT);
  transpose_to_f16<<<(D_HID * D_OUT + T - 1) / T, T, 0, stream>>>(W2r, w2rt, D_HID, D_OUT);

  // ---- layer 1 ----
  scatter_add<<<(N_EDGES * D_IN) / T, T, 0, stream>>>(x, src, dst, agg1, N_EDGES, 7);
  mean_to_f16_v4<<<(N_NODES * D_IN / 4 + T - 1) / T, T, 0, stream>>>(
      agg1, cnt, mean1h, N_NODES, 7);
  dim3 g1(D_HID / 128, N_NODES / 16);
  sage_gemm<D_IN, 0><<<g1, 256, 0, stream>>>(mean1h, w1lt, xh, w1rt, b1,
                                             x1, x1h, D_HID);

  // ---- layer 2 ----
  scatter_add<<<(N_EDGES * D_HID) / T, T, 0, stream>>>(x1, src, dst, agg2, N_EDGES, 8);
  mean_to_f16_v4<<<(N_NODES * D_HID / 4 + T - 1) / T, T, 0, stream>>>(
      agg2, cnt, mean2h, N_NODES, 8);
  dim3 g2(D_OUT / 128, N_NODES / 16);
  sage_gemm<D_HID, 1><<<g2, 256, 0, stream>>>(mean2h, w2lt, x1h, w2rt, b2,
                                              (float*)d_out, (_Float16*)nullptr,
                                              D_OUT);
}
